// RSSM_37426345017513
// MI455X (gfx1250) — compile-verified, matrix operand order checked by reading, not run on hardware
//
#include <hip/hip_runtime.h>
#include <hip/hip_bf16.h>

typedef __bf16 bf16_t;
typedef bf16_t v16bf __attribute__((ext_vector_type(16)));
typedef float  v8f   __attribute__((ext_vector_type(8)));

#define DEV static __device__ __forceinline__

constexpr int S    = 64;    // hidden state
constexpr int ACTD = 32;    // action dim
constexpr int HID  = 128;   // decoder hidden
constexpr int OBSD = 256;   // obs dim
constexpr int TT   = 128;   // horizon
constexpr int BB   = 2048;  // batch
constexpr int WAVES = 2;    // waves per block, each owns a 16-row batch tile

constexpr int FRAG_ELEMS = 512; // one 32x16 bf16 B-fragment (1 KB)

// Fragment table (frag index = base + nt*KT + kt); weights stored as W^T (K x N).
// LDS-resident fragments first (96 frags = 96 KB), dec_w2 (64 frags) last, kept in global/L2.
constexpr int F_WIH = 0;            // (K=96 ->3kt) x (N=192 ->12nt) = 36
constexpr int F_WHH = F_WIH + 36;   // (64->2) x (192->12) = 24
constexpr int F_D1  = F_WHH + 24;   // (64->2) x (128->8)  = 16
constexpr int F_RW1 = F_D1  + 16;   // (64->2) x (64->4)   = 8
constexpr int F_RW2 = F_RW1 + 8;    // (64->2) x (1->1)    = 2
constexpr int F_SF1 = F_RW2 + 2;    // 8
constexpr int F_SF2 = F_SF1 + 8;    // 2
constexpr int NFRAG_LDS = F_SF2 + 2;   // 96
constexpr int F_D2  = NFRAG_LDS;    // (128->4) x (256->16)= 64
constexpr int NFRAG = F_D2 + 64;    // 160 frags = 160 KB bf16

// ---------------------------------------------------------------- pack kernel
// Convert f32 weights -> bf16 WMMA B-fragments in workspace.
// B-fragment layout (16x16x32 bf16): lane l, element j:
//   N = nt*16 + (l & 15);  K = kt*32 + (l >> 4)*16 + j
__global__ void pack_weights(const float* __restrict__ wih, const float* __restrict__ whh,
                             const float* __restrict__ d1,  const float* __restrict__ d2,
                             const float* __restrict__ rw1, const float* __restrict__ rw2,
                             const float* __restrict__ sf1, const float* __restrict__ sf2,
                             bf16_t* __restrict__ out) {
    int gid = blockIdx.x * blockDim.x + threadIdx.x;
    if (gid >= NFRAG * FRAG_ELEMS) return;
    int frag = gid >> 9;
    int pos  = gid & 511;
    int lane = pos >> 4, j = pos & 15;
    int nloc = lane & 15, lh = lane >> 4;

    const float* W; int Kin, Nout, KT, base;
    if      (frag < F_WHH) { W = wih; Kin = 96;  Nout = 192; KT = 3; base = F_WIH; }
    else if (frag < F_D1 ) { W = whh; Kin = 64;  Nout = 192; KT = 2; base = F_WHH; }
    else if (frag < F_RW1) { W = d1;  Kin = 64;  Nout = 128; KT = 2; base = F_D1;  }
    else if (frag < F_RW2) { W = rw1; Kin = 64;  Nout = 64;  KT = 2; base = F_RW1; }
    else if (frag < F_SF1) { W = rw2; Kin = 64;  Nout = 1;   KT = 2; base = F_RW2; }
    else if (frag < F_SF2) { W = sf1; Kin = 64;  Nout = 64;  KT = 2; base = F_SF1; }
    else if (frag < F_D2 ) { W = sf2; Kin = 64;  Nout = 1;   KT = 2; base = F_SF2; }
    else                   { W = d2;  Kin = 128; Nout = 256; KT = 4; base = F_D2;  }

    int fl = frag - base;
    int nt = fl / KT, kt = fl % KT;
    int n = nt * 16 + nloc;
    int k = kt * 32 + lh * 16 + j;
    float v = (n < Nout && k < Kin) ? W[n * Kin + k] : 0.0f; // W is (Nout,Kin) row-major
    out[gid] = (bf16_t)v;
}

// ---------------------------------------------------------------- helpers
DEV v8f wmma_bf16(v16bf a, v16bf b, v8f c) {
    return __builtin_amdgcn_wmma_f32_16x16x32_bf16(false, a, false, b, (short)0, c, false, false);
}

// Opaque zero: regenerated every loop iteration so fragment loads are
// loop-variant -> LICM cannot hoist them into (spilled) registers.
DEV int opaque_zero() {
    int x = 0;
    asm volatile("" : "+s"(x));
    return x;
}

// A-fragment from row-major bf16 LDS tile (stride elems): lane l -> M=l&15, K half = l>>4
DEV v16bf lds_a(const bf16_t* base, int stride, int kt, int lane) {
    return *(const v16bf*)(base + (lane & 15) * stride + kt * 32 + (lane >> 4) * 16);
}
// B-fragment from LDS weight cache (32 contiguous bytes per lane)
DEV v16bf lds_b(const bf16_t* wl, int frag, int lane, int lo) {
    return *(const v16bf*)(wl + frag * FRAG_ELEMS + lane * 16 + lo);
}
// B-fragment from packed global workspace (coalesced 1 KB burst, L2-hot)
DEV v16bf gmem_b(const bf16_t* wf, int frag, int lane, int lo) {
    return *(const v16bf*)(wf + (size_t)(frag * FRAG_ELEMS + lane * 16 + lo));
}
DEV float sigm_fast(float x) {  // rcp(1 + exp2(-x*log2e)) : v_exp_f32 + v_rcp_f32
    return __builtin_amdgcn_rcpf(1.0f + __builtin_amdgcn_exp2f(-1.4426950408889634f * x));
}
DEV float tanh_fast(float x) {  // 1 - 2*rcp(1 + exp2(2x*log2e))
    return 1.0f - 2.0f * __builtin_amdgcn_rcpf(1.0f + __builtin_amdgcn_exp2f(2.8853900817779268f * x));
}

// ---------------------------------------------------------------- main kernel
__global__ __launch_bounds__(32 * WAVES)
void rssm_rollout(const float* __restrict__ h0, const float* __restrict__ actions,
                  const bf16_t* __restrict__ wf,
                  const float* __restrict__ bih, const float* __restrict__ bhh,
                  const float* __restrict__ bd1g, const float* __restrict__ bd2g,
                  const float* __restrict__ br1g, const float* __restrict__ br2g,
                  const float* __restrict__ bs1g, const float* __restrict__ bs2g,
                  float* __restrict__ out_states, float* __restrict__ out_obs,
                  float* __restrict__ out_rew,    float* __restrict__ out_saf) {
    __shared__ __align__(32) bf16_t wlds[NFRAG_LDS * FRAG_ELEMS];  // 96 KB weight cache
    __shared__ __align__(32) bf16_t xbuf[WAVES][2][16 * 96];       // x = [h | a], ping-pong
    __shared__ __align__(32) bf16_t midbuf[WAVES][16 * HID];       // decoder hidden
    __shared__ __align__(32) bf16_t hdbuf[WAVES][16 * S];          // rew/saf hidden (reused)

    const int lane = threadIdx.x & 31;
    const int w    = threadIdx.x >> 5;
    const int ln   = lane & 15;     // N / M index within tile
    const int lh   = lane >> 4;     // half selector
    const int b0   = (blockIdx.x * WAVES + w) * 16;

    // ---- cooperative stage: first 96 fragments (all weights except dec_w2) -> LDS
    {
        const uint4* src = (const uint4*)wf;
        uint4* dst = (uint4*)wlds;
        constexpr int NV4 = NFRAG_LDS * FRAG_ELEMS * (int)sizeof(bf16_t) / 16;
        for (int i = threadIdx.x; i < NV4; i += 32 * WAVES) dst[i] = src[i];
    }
    __syncthreads();

    bf16_t* x0 = &xbuf[w][0][0];
    bf16_t* x1 = &xbuf[w][1][0];
    bf16_t* mid = &midbuf[w][0];
    bf16_t* hd  = &hdbuf[w][0];

    // ---- persistent hidden state in registers (C-layout) + preload biases
    float hreg[4][8];
    float br[4], bz[4], bin[4], bhn[4], br1[4], bs1[4], bd1[8], bd2[16];
#pragma unroll
    for (int ct = 0; ct < 4; ++ct) {
        int col = ct * 16 + ln;
        br[ct]  = bih[col]       + bhh[col];
        bz[ct]  = bih[64 + col]  + bhh[64 + col];
        bin[ct] = bih[128 + col];
        bhn[ct] = bhh[128 + col];
        br1[ct] = br1g[col];
        bs1[ct] = bs1g[col];
#pragma unroll
        for (int v = 0; v < 8; ++v) {
            int m = v + 8 * lh;
            float hv = h0[(size_t)(b0 + m) * S + col];
            hreg[ct][v] = hv;
            out_states[(size_t)(b0 + m) * (TT + 1) * S + col] = hv; // states[:,0,:]
            x0[m * 96 + col] = (bf16_t)hv;
        }
    }
#pragma unroll
    for (int nt = 0; nt < 8; ++nt)  bd1[nt] = bd1g[nt * 16 + ln];
#pragma unroll
    for (int nt = 0; nt < 16; ++nt) bd2[nt] = bd2g[nt * 16 + ln];
    const float br2 = br2g[0], bs2 = bs2g[0];

    // action staging: lane covers row = lane>>1, 16 cols starting at (lane&1)*16
    const int arow = lane >> 1, acb = (lane & 1) * 16;
    const float* abase = actions + (size_t)(b0 + arow) * TT * ACTD + acb;

#pragma unroll 1
    for (int t = 0; t < TT; ++t) {
        const int lo = opaque_zero();     // defeat LICM on fragment loads
        bf16_t* xr = (t & 1) ? x1 : x0;   // holds h_{t-1}
        bf16_t* xw = (t & 1) ? x0 : x1;   // will hold h_t

        // ---- stage a_t (bf16) into read buffer cols 64..95
        const float* ap = abase + (size_t)t * ACTD;
        const float4 a4a = *(const float4*)(ap + 0);
        const float4 a4b = *(const float4*)(ap + 4);
        const float4 a4c = *(const float4*)(ap + 8);
        const float4 a4d = *(const float4*)(ap + 12);
        bf16_t* xa = xr + arow * 96 + 64 + acb;
        xa[0]=(bf16_t)a4a.x; xa[1]=(bf16_t)a4a.y; xa[2]=(bf16_t)a4a.z; xa[3]=(bf16_t)a4a.w;
        xa[4]=(bf16_t)a4b.x; xa[5]=(bf16_t)a4b.y; xa[6]=(bf16_t)a4b.z; xa[7]=(bf16_t)a4b.w;
        xa[8]=(bf16_t)a4c.x; xa[9]=(bf16_t)a4c.y; xa[10]=(bf16_t)a4c.z; xa[11]=(bf16_t)a4c.w;
        xa[12]=(bf16_t)a4d.x; xa[13]=(bf16_t)a4d.y; xa[14]=(bf16_t)a4d.z; xa[15]=(bf16_t)a4d.w;
        if (t + 1 < TT) __builtin_prefetch(ap + ACTD, 0, 0);

        // ---- GRU: per S n-tile ct, compute r/z/n gates via WMMA
#pragma unroll
        for (int ct = 0; ct < 4; ++ct) {
            v8f ar = {}, az = {}, ain = {}, ahn = {};
#pragma unroll
            for (int kt = 0; kt < 3; ++kt) {                 // gi = x @ W_ih^T  (K=96)
                v16bf a = lds_a(xr, 96, kt, lane);
                ar  = wmma_bf16(a, lds_b(wlds, F_WIH + (ct     ) * 3 + kt, lane, lo), ar);
                az  = wmma_bf16(a, lds_b(wlds, F_WIH + (ct + 4 ) * 3 + kt, lane, lo), az);
                ain = wmma_bf16(a, lds_b(wlds, F_WIH + (ct + 8 ) * 3 + kt, lane, lo), ain);
            }
#pragma unroll
            for (int kt = 0; kt < 2; ++kt) {                 // gh = h @ W_hh^T  (K=64)
                v16bf a = lds_a(xr, 96, kt, lane);           // h = cols 0..63 of x
                ar  = wmma_bf16(a, lds_b(wlds, F_WHH + (ct     ) * 2 + kt, lane, lo), ar);
                az  = wmma_bf16(a, lds_b(wlds, F_WHH + (ct + 4 ) * 2 + kt, lane, lo), az);
                ahn = wmma_bf16(a, lds_b(wlds, F_WHH + (ct + 8 ) * 2 + kt, lane, lo), ahn);
            }
            int col = ct * 16 + ln;
#pragma unroll
            for (int v = 0; v < 8; ++v) {
                float r  = sigm_fast(ar[v] + br[ct]);
                float z  = sigm_fast(az[v] + bz[ct]);
                float nn = tanh_fast(ain[v] + bin[ct] + r * (ahn[v] + bhn[ct]));
                float h2 = (1.0f - z) * nn + z * hreg[ct][v];
                hreg[ct][v] = h2;
                int m = v + 8 * lh;
                xw[m * 96 + col] = (bf16_t)h2;
                out_states[(size_t)(b0 + m) * (TT + 1) * S + (size_t)(t + 1) * S + col] = h2;
            }
        }

        // ---- decoder: mid = relu(h2 @ dec_w1^T + b1)   (16x128)
#pragma unroll
        for (int nt = 0; nt < 8; ++nt) {
            v8f acc = {};
#pragma unroll
            for (int kt = 0; kt < 2; ++kt)
                acc = wmma_bf16(lds_a(xw, 96, kt, lane), lds_b(wlds, F_D1 + nt * 2 + kt, lane, lo), acc);
            int col = nt * 16 + ln;
#pragma unroll
            for (int v = 0; v < 8; ++v)
                mid[(v + 8 * lh) * HID + col] = (bf16_t)fmaxf(acc[v] + bd1[nt], 0.0f);
        }
        // ---- obs = mid @ dec_w2^T + b2   (16x256) -> global ; dec_w2 frags stream from L2
#pragma unroll
        for (int nt = 0; nt < 16; ++nt) {
            v8f acc = {};
#pragma unroll
            for (int kt = 0; kt < 4; ++kt)
                acc = wmma_bf16(lds_a(mid, HID, kt, lane), gmem_b(wf, F_D2 + nt * 4 + kt, lane, lo), acc);
            int col = nt * 16 + ln;
#pragma unroll
            for (int v = 0; v < 8; ++v)
                out_obs[(size_t)(b0 + v + 8 * lh) * TT * OBSD + (size_t)t * OBSD + col] = acc[v] + bd2[nt];
        }

        // ---- reward head: relu(h2 @ rew_w1^T + b1) @ rew_w2^T + b2
#pragma unroll
        for (int nt = 0; nt < 4; ++nt) {
            v8f acc = {};
#pragma unroll
            for (int kt = 0; kt < 2; ++kt)
                acc = wmma_bf16(lds_a(xw, 96, kt, lane), lds_b(wlds, F_RW1 + nt * 2 + kt, lane, lo), acc);
            int col = nt * 16 + ln;
#pragma unroll
            for (int v = 0; v < 8; ++v)
                hd[(v + 8 * lh) * S + col] = (bf16_t)fmaxf(acc[v] + br1[nt], 0.0f);
        }
        {
            v8f acc = {};
#pragma unroll
            for (int kt = 0; kt < 2; ++kt)
                acc = wmma_bf16(lds_a(hd, S, kt, lane), lds_b(wlds, F_RW2 + kt, lane, lo), acc);
            if (ln == 0) {
#pragma unroll
                for (int v = 0; v < 8; ++v)
                    out_rew[(size_t)(b0 + v + 8 * lh) * TT + t] = acc[v] + br2;
            }
        }
        // ---- safety head (reuses hd buffer; LDS is in-order within a wave)
#pragma unroll
        for (int nt = 0; nt < 4; ++nt) {
            v8f acc = {};
#pragma unroll
            for (int kt = 0; kt < 2; ++kt)
                acc = wmma_bf16(lds_a(xw, 96, kt, lane), lds_b(wlds, F_SF1 + nt * 2 + kt, lane, lo), acc);
            int col = nt * 16 + ln;
#pragma unroll
            for (int v = 0; v < 8; ++v)
                hd[(v + 8 * lh) * S + col] = (bf16_t)fmaxf(acc[v] + bs1[nt], 0.0f);
        }
        {
            v8f acc = {};
#pragma unroll
            for (int kt = 0; kt < 2; ++kt)
                acc = wmma_bf16(lds_a(hd, S, kt, lane), lds_b(wlds, F_SF2 + kt, lane, lo), acc);
            if (ln == 0) {
#pragma unroll
                for (int v = 0; v < 8; ++v)
                    out_saf[(size_t)(b0 + v + 8 * lh) * TT + t] = acc[v] + bs2;
            }
        }
    }
}

// ---------------------------------------------------------------- launch
extern "C" void kernel_launch(void* const* d_in, const int* in_sizes, int n_in,
                              void* d_out, int out_size, void* d_ws, size_t ws_size,
                              hipStream_t stream) {
    const float* h0  = (const float*)d_in[0];
    const float* act = (const float*)d_in[1];
    const float* wih = (const float*)d_in[2];
    const float* whh = (const float*)d_in[3];
    const float* bih = (const float*)d_in[4];
    const float* bhh = (const float*)d_in[5];
    const float* d1w = (const float*)d_in[6];
    const float* d1b = (const float*)d_in[7];
    const float* d2w = (const float*)d_in[8];
    const float* d2b = (const float*)d_in[9];
    const float* r1w = (const float*)d_in[10];
    const float* r1b = (const float*)d_in[11];
    const float* r2w = (const float*)d_in[12];
    const float* r2b = (const float*)d_in[13];
    const float* s1w = (const float*)d_in[14];
    const float* s1b = (const float*)d_in[15];
    const float* s2w = (const float*)d_in[16];
    const float* s2b = (const float*)d_in[17];
    (void)in_sizes; (void)n_in; (void)out_size; (void)ws_size;

    bf16_t* wf = (bf16_t*)d_ws;
    float* out = (float*)d_out;
    float* out_states = out;
    float* out_obs = out_states + (size_t)BB * (TT + 1) * S;
    float* out_rew = out_obs + (size_t)BB * TT * OBSD;
    float* out_saf = out_rew + (size_t)BB * TT;

    pack_weights<<<(NFRAG * FRAG_ELEMS + 255) / 256, 256, 0, stream>>>(
        wih, whh, d1w, d2w, r1w, r2w, s1w, s2w, wf);

    rssm_rollout<<<BB / (16 * WAVES), 32 * WAVES, 0, stream>>>(
        h0, act, wf, bih, bhh, d1b, d2b, r1b, r2b, s1b, s2b,
        out_states, out_obs, out_rew, out_saf);
}
